// RelationalBBoxNetworkEncoder_27006754357648
// MI455X (gfx1250) — matrix operand order repaired and checked by reading, not compiled, not run on hardware
//
#include <hip/hip_runtime.h>

// ---------------------------------------------------------------------------
// RelationalBBoxNetworkEncoder for MI455X (gfx1250, wave32, WMMA).
// All heavy math runs as f16 WMMA (v_wmma_f32_16x16x32_f16, f32 accum).
// g1 is algebraically factored: h1[a,b] = ReLU(u[b] + v[a] + b1).
// Weights converted once to f16 [N][K] row-major so B fragments are
// two 16-contiguous-byte per-lane global loads (L2-resident, 192MB).
// conv3x3 A-tiles stream into LDS via async-to-LDS DMA (double buffered)
// when the toolchain exposes the gfx1250 async builtins.
// ---------------------------------------------------------------------------

typedef __attribute__((ext_vector_type(16))) _Float16 v16h;
typedef __attribute__((ext_vector_type(8)))  _Float16 v8h;
typedef __attribute__((ext_vector_type(8)))  float    v8f;

#if __has_builtin(__builtin_amdgcn_global_load_async_to_lds_b128) && \
    __has_builtin(__builtin_amdgcn_s_wait_asynccnt)
#define USE_ASYNC_LDS 1
// Builtin signature (from hipcc diagnostic): param0 is a non-const pointer to
// 'int __attribute__((vector_size(16)))' in address space 1 (global).
typedef int vi4 __attribute__((vector_size(16)));
typedef __attribute__((address_space(1))) vi4* gp_vi4;
typedef __attribute__((address_space(3))) vi4* lp_vi4;
#else
#define USE_ASYNC_LDS 0
#endif

union FragU { v16h v; v8h h[2]; };

// A fragment: 16x32 (MxK) tile, row-major, row stride ld (halves).
// ISA layout: lane l (m=l&15, hi=l>>4) holds k = hi*8+j (j<8) and 16+hi*8+j.
__device__ __forceinline__ v16h load_a_frag(const _Float16* tile, int ld) {
  const int l  = threadIdx.x & 31;
  const int m  = l & 15;
  const int hi = l >> 4;
  const _Float16* p = tile + (size_t)m * ld + hi * 8;
  FragU f;
  f.h[0] = *(const v8h*)(p);
  f.h[1] = *(const v8h*)(p + 16);
  return f.v;
}

// B fragment: 32x16 (KxN) tile whose storage is [N][K] row-major (ld = K stride).
// ISA layout: lane l (n=l&15, hi=l>>4) holds k = hi*8+j (j<8) and 16+hi*8+j.
__device__ __forceinline__ v16h load_b_frag(const _Float16* w, int ld) {
  const int l  = threadIdx.x & 31;
  const int n  = l & 15;
  const int hi = l >> 4;
  const _Float16* p = w + (size_t)n * ld + hi * 8;
  FragU f;
  f.h[0] = *(const v8h*)(p);
  f.h[1] = *(const v8h*)(p + 16);
  return f.v;
}

__device__ __forceinline__ v8f wmma16(v16h a, v16h b, v8f c) {
  return __builtin_amdgcn_wmma_f32_16x16x32_f16(false, a, false, b, (short)0, c,
                                                false, false);
}

// ---------------------------------------------------------------------------
// Weight / activation conversion kernels (f32 -> f16, with layout permutes)
// ---------------------------------------------------------------------------

__global__ void k_f32_to_f16(const float* __restrict__ src, _Float16* __restrict__ dst, int n) {
  int i = blockIdx.x * blockDim.x + threadIdx.x;
  if (i < n) dst[i] = (_Float16)src[i];
}

// conv_w [256][1024][3][3] -> f16 [o][kk][c] (kk = dy*3+dx), K order = kk*1024+c
__global__ void k_cvt_convw(const float* __restrict__ src, _Float16* __restrict__ dst) {
  int i = blockIdx.x * blockDim.x + threadIdx.x;
  if (i >= 256 * 9216) return;
  int o = i / 9216, rem = i % 9216;
  int kk = rem / 1024, c = rem % 1024;
  dst[i] = (_Float16)src[(size_t)(o * 1024 + c) * 9 + kk];
}

// fc_w [1024][256*49] (K=c*49+s) -> f16 [o][s*256+c] to match conv-out [ri][s][c]
__global__ void k_cvt_fcw(const float* __restrict__ src, _Float16* __restrict__ dst) {
  int i = blockIdx.x * blockDim.x + threadIdx.x;
  if (i >= 1024 * 12544) return;
  int o = i / 12544, rem = i % 12544;
  int s = rem / 256, c = rem % 256;
  dst[i] = (_Float16)src[(size_t)o * 12544 + c * 49 + s];
}

// im_feat [8][2048][256] NCHW -> f16 [8][256(pix)][2048(c)] (NHWC) for B frags
__global__ void k_cvt_imf(const float* __restrict__ src, _Float16* __restrict__ dst) {
  int i = blockIdx.x * blockDim.x + threadIdx.x;
  if (i >= 8 * 256 * 2048) return;
  int b = i / (256 * 2048), rem = i % (256 * 2048);
  int p = rem / 2048, c = rem % 2048;
  dst[i] = (_Float16)src[((size_t)b * 2048 + c) * 256 + p];
}

// Zero the 32 border cells of each roi's padded 9x9 grid: [512][81][1024] f16.
// grid (512, 32), block 128 (8 halves per thread).
__global__ void k_zero_border(_Float16* __restrict__ rbuf) {
  const int ri = blockIdx.x;
  const int j = blockIdx.y;
  int y, x;
  if (j < 9)       { y = 0;      x = j;      }
  else if (j < 18) { y = 8;      x = j - 9;  }
  else if (j < 25) { y = j - 17; x = 0;      }   // y = 1..7
  else             { y = j - 24; x = 8;      }   // y = 1..7
  v8h z = {};
  *(v8h*)(rbuf + ((size_t)ri * 81 + y * 9 + x) * 1024 + threadIdx.x * 8) = z;
}

// ---------------------------------------------------------------------------
// K1: 1x1 projection GEMM. M=1024 (out ch), N=256 (pixels), K=2048, per image.
// grid (16, 4, 8), block 256 (8 waves; wave -> msub = w&3, nsubs = (w>>2)*2+{0,1})
// ---------------------------------------------------------------------------
__global__ void __launch_bounds__(256) k_proj(const _Float16* __restrict__ projw,
                                              const _Float16* __restrict__ imf,
                                              const float* __restrict__ projb,
                                              float* __restrict__ xout) {
  const int img = blockIdx.z;
  const int m0 = blockIdx.x * 64;
  const int n0 = blockIdx.y * 64;
  const int w = threadIdx.x >> 5;
  const int msub = w & 3;
  const int nsb = (w >> 2) * 2;
  const _Float16* A = projw + (size_t)(m0 + msub * 16) * 2048;
  const _Float16* B = imf + (size_t)img * 256 * 2048 + (size_t)n0 * 2048;
  v8f acc0 = {}, acc1 = {};
  for (int k = 0; k < 2048; k += 32) {
    v16h a = load_a_frag(A + k, 2048);
    v16h b0 = load_b_frag(B + (size_t)(nsb * 16) * 2048 + k, 2048);
    v16h b1 = load_b_frag(B + (size_t)(nsb * 16 + 16) * 2048 + k, 2048);
    acc0 = wmma16(a, b0, acc0);
    acc1 = wmma16(a, b1, acc1);
  }
  const int l = threadIdx.x & 31, n = l & 15, hi = l >> 4;
  float* O = xout + (size_t)img * 1024 * 256 + (size_t)(m0 + msub * 16) * 256 + n0;
  for (int r = 0; r < 8; ++r) {
    int mm = r + hi * 8;
    float bias = projb[m0 + msub * 16 + mm];
    O[(size_t)mm * 256 + nsb * 16 + n] = acc0[r] + bias;
    O[(size_t)mm * 256 + nsb * 16 + 16 + n] = acc1[r] + bias;
  }
}

// ---------------------------------------------------------------------------
// K2: ROIAlignV2 (aligned=True), 2x2 samples/bin, average. Output channel-last
// f16 [512][9][9][1024] (zero-padded halo; interior cell = (y+1, x+1)).
// grid (512, 49), block 256 (4 channels/thread).
// ---------------------------------------------------------------------------
__global__ void __launch_bounds__(256) k_roialign(const float* __restrict__ x,
                                                  const float* __restrict__ boxes,
                                                  _Float16* __restrict__ r) {
  const int ri = blockIdx.x;
  const int bin = blockIdx.y;
  const int img = ri >> 6;
  const float scale = 1.0f / 32.0f;
  const float* bx = boxes + ri * 4;
  float rx1 = bx[0] * scale - 0.5f;
  float ry1 = bx[1] * scale - 0.5f;
  float bw = (bx[2] - bx[0]) * scale * (1.0f / 7.0f);
  float bh = (bx[3] - bx[1]) * scale * (1.0f / 7.0f);
  int byi = bin / 7, bxi = bin - byi * 7;
  int c = threadIdx.x;
  float acc[4] = {0.f, 0.f, 0.f, 0.f};
  const float* base = x + (size_t)img * 1024 * 256;
  for (int sy = 0; sy < 2; ++sy)
    for (int sx = 0; sx < 2; ++sx) {
      float ys = ry1 + bh * ((float)(byi * 2 + sy) + 0.5f) * 0.5f;
      float xs = rx1 + bw * ((float)(bxi * 2 + sx) + 0.5f) * 0.5f;
      bool vy = (ys >= -1.0f) && (ys <= 16.0f);
      bool vx = (xs >= -1.0f) && (xs <= 16.0f);
      float yc = fmaxf(ys, 0.0f), xc = fmaxf(xs, 0.0f);
      int yl = (int)floorf(yc), xl = (int)floorf(xc);
      if (yl >= 15) { yl = 15; yc = 15.0f; }
      if (xl >= 15) { xl = 15; xc = 15.0f; }
      int yh = (yl < 15) ? yl + 1 : yl;
      int xh = (xl < 15) ? xl + 1 : xl;
      float wy = yc - (float)yl, wx = xc - (float)xl;
      float msk = (vy && vx) ? 0.25f : 0.0f;
      float w00 = (1.f - wy) * (1.f - wx) * msk, w01 = (1.f - wy) * wx * msk;
      float w10 = wy * (1.f - wx) * msk, w11 = wy * wx * msk;
      for (int j = 0; j < 4; ++j) {
        const float* ch = base + (size_t)(c + j * 256) * 256;
        acc[j] += ch[yl * 16 + xl] * w00 + ch[yl * 16 + xh] * w01 +
                  ch[yh * 16 + xl] * w10 + ch[yh * 16 + xh] * w11;
      }
    }
  _Float16* dst = r + ((size_t)ri * 81 + (byi + 1) * 9 + (bxi + 1)) * 1024;
  for (int j = 0; j < 4; ++j) dst[c + j * 256] = (_Float16)acc[j];
}

// ---------------------------------------------------------------------------
// K3: conv3x3 as implicit GEMM per roi. M=64 (49 spatial + pad), N=256, K=9216.
// A tiles stream from the zero-padded ROI buffer into double-buffered LDS via
// async-to-LDS DMA (ASYNCcnt) when available; B frags direct from global f16.
// Fused BN(eval)+ReLU. Output f16 [512][49][256]. grid 512, block 256.
// ---------------------------------------------------------------------------
__device__ __forceinline__ const _Float16* conv3_src(const _Float16* rbuf, int ri,
                                                     int s, int m, int kl) {
  const int plane = s >> 5;            // 9 planes (dy,dx), 32 k-steps each
  const int cbase = (s & 31) * 32;
  const int dy = plane / 3, dx = plane % 3;  // padded offsets 0..2
  // rows >= 49 read the (0,0) border cell (pre-zeroed halo)
  size_t cell = (m < 49) ? ((size_t)ri * 81 + (m / 7 + dy) * 9 + (m % 7 + dx))
                         : ((size_t)ri * 81);
  return rbuf + cell * 1024 + cbase + kl;
}

__global__ void __launch_bounds__(256) k_conv3(const _Float16* __restrict__ rbuf,
                                               const _Float16* __restrict__ convw,
                                               const float* __restrict__ convb,
                                               const float* __restrict__ bng,
                                               const float* __restrict__ bnb,
                                               const float* __restrict__ bnm,
                                               const float* __restrict__ bnv,
                                               _Float16* __restrict__ cout) {
  __shared__ _Float16 amat[2][64 * 32];
  const int ri = blockIdx.x;
  const int t = threadIdx.x;
  const int w = t >> 5;
  const int n0 = w * 32;               // this wave's 32 output channels
  const int sm = t >> 2;               // staging row 0..63
  const int skl = (t & 3) * 8;         // staging k-chunk (8 halves = 16B)

  auto stage = [&](int s, int buf) {
    const _Float16* src = conv3_src(rbuf, ri, s, sm, skl);
    _Float16* dst = &amat[buf][sm * 32 + skl];
#if USE_ASYNC_LDS
    __builtin_amdgcn_global_load_async_to_lds_b128((gp_vi4)src, (lp_vi4)dst, 0, 0);
#else
    *(v8h*)dst = *(const v8h*)src;
#endif
  };

  v8f acc[4][2] = {};
  stage(0, 0);                         // prologue
  for (int s = 0; s < 287; ++s) {
    stage(s + 1, (s + 1) & 1);         // issue DMA for next tile
#if USE_ASYNC_LDS
    __builtin_amdgcn_s_wait_asynccnt(1);   // drain tile s, keep s+1 in flight
#endif
    __syncthreads();
    const _Float16* Brow = convw + (size_t)n0 * 9216 + s * 32;
    v16h b0 = load_b_frag(Brow, 9216);
    v16h b1 = load_b_frag(Brow + (size_t)16 * 9216, 9216);
    const _Float16* at = amat[s & 1];
    for (int ms = 0; ms < 4; ++ms) {
      v16h a = load_a_frag(at + ms * 16 * 32, 32);
      acc[ms][0] = wmma16(a, b0, acc[ms][0]);
      acc[ms][1] = wmma16(a, b1, acc[ms][1]);
    }
    __syncthreads();                   // reads of buf(s) done before re-stage
  }
  {                                    // final step s = 287
#if USE_ASYNC_LDS
    __builtin_amdgcn_s_wait_asynccnt(0);
#endif
    __syncthreads();
    const _Float16* Brow = convw + (size_t)n0 * 9216 + 287 * 32;
    v16h b0 = load_b_frag(Brow, 9216);
    v16h b1 = load_b_frag(Brow + (size_t)16 * 9216, 9216);
    const _Float16* at = amat[287 & 1];
    for (int ms = 0; ms < 4; ++ms) {
      v16h a = load_a_frag(at + ms * 16 * 32, 32);
      acc[ms][0] = wmma16(a, b0, acc[ms][0]);
      acc[ms][1] = wmma16(a, b1, acc[ms][1]);
    }
  }
  const int l = t & 31, n = l & 15, hi = l >> 4;
  for (int ns = 0; ns < 2; ++ns) {
    int o = n0 + ns * 16 + n;
    float inv = rsqrtf(bnv[o] + 1e-5f);
    float sc = bng[o] * inv;
    float off = (convb[o] - bnm[o]) * sc + bnb[o];
    for (int ms = 0; ms < 4; ++ms)
      for (int r = 0; r < 8; ++r) {
        int m = ms * 16 + r + hi * 8;
        if (m < 49) {
          float vv = fmaxf(acc[ms][ns][r] * sc + off, 0.0f);
          cout[((size_t)ri * 49 + m) * 256 + o] = (_Float16)vv;
        }
      }
  }
}

// ---------------------------------------------------------------------------
// K4: FC GEMM + bias + ReLU. M=512, N=1024, K=12544. grid (8,16), block 256.
// ---------------------------------------------------------------------------
__global__ void __launch_bounds__(256) k_fc(const _Float16* __restrict__ Abase,
                                            const _Float16* __restrict__ W,
                                            const float* __restrict__ bias,
                                            float* __restrict__ out) {
  const int m0 = blockIdx.x * 64, n0 = blockIdx.y * 64;
  const int w = threadIdx.x >> 5, msub = w & 3, nsb = (w >> 2) * 2;
  const _Float16* A = Abase + (size_t)(m0 + msub * 16) * 12544;
  const _Float16* B0 = W + (size_t)(n0 + nsb * 16) * 12544;
  const _Float16* B1 = W + (size_t)(n0 + nsb * 16 + 16) * 12544;
  v8f acc0 = {}, acc1 = {};
  for (int k = 0; k < 12544; k += 32) {
    __builtin_prefetch(B0 + k + 512, 0, 1);
    v16h a = load_a_frag(A + k, 12544);
    v16h b0 = load_b_frag(B0 + k, 12544);
    v16h b1 = load_b_frag(B1 + k, 12544);
    acc0 = wmma16(a, b0, acc0);
    acc1 = wmma16(a, b1, acc1);
  }
  const int l = threadIdx.x & 31, n = l & 15, hi = l >> 4;
  float* O = out + (size_t)(m0 + msub * 16) * 1024 + n0;
  for (int r = 0; r < 8; ++r) {
    int mm = r + hi * 8;
    int c0 = nsb * 16 + n, c1 = c0 + 16;
    O[(size_t)mm * 1024 + c0] = fmaxf(acc0[r] + bias[n0 + c0], 0.0f);
    O[(size_t)mm * 1024 + c1] = fmaxf(acc1[r] + bias[n0 + c1], 0.0f);
  }
}

// ---------------------------------------------------------------------------
// K5: feats = [LN1(fc row) | LN2(box branch)] -> f16 [512][1280]. grid 512.
// ---------------------------------------------------------------------------
__global__ void __launch_bounds__(256) k_feats(const float* __restrict__ fcout,
                                               const float* __restrict__ ln1g,
                                               const float* __restrict__ ln1b,
                                               const float* __restrict__ nboxes,
                                               const float* __restrict__ boxw,
                                               const float* __restrict__ boxb,
                                               const float* __restrict__ ln2g,
                                               const float* __restrict__ ln2b,
                                               _Float16* __restrict__ feats) {
  __shared__ float red[256];
  const int ri = blockIdx.x, t = threadIdx.x;
  const float* row = fcout + (size_t)ri * 1024;
  float vals[4];
  float s = 0.0f;
  for (int j = 0; j < 4; ++j) { vals[j] = row[t + j * 256]; s += vals[j]; }
  red[t] = s; __syncthreads();
  for (int st = 128; st; st >>= 1) { if (t < st) red[t] += red[t + st]; __syncthreads(); }
  float mu = red[0] * (1.0f / 1024.0f); __syncthreads();
  s = 0.0f;
  for (int j = 0; j < 4; ++j) { float d = vals[j] - mu; s += d * d; }
  red[t] = s; __syncthreads();
  for (int st = 128; st; st >>= 1) { if (t < st) red[t] += red[t + st]; __syncthreads(); }
  float rstd = rsqrtf(red[0] * (1.0f / 1024.0f) + 1e-5f); __syncthreads();
  _Float16* frow = feats + (size_t)ri * 1280;
  for (int j = 0; j < 4; ++j) {
    int i = t + j * 256;
    frow[i] = (_Float16)((vals[j] - mu) * rstd * ln1g[i] + ln1b[i]);
  }
  // box branch: nb (4) -> 256, then LN over 256
  float nb0 = nboxes[ri * 4 + 0] * 2.0f - 1.0f;
  float nb1 = nboxes[ri * 4 + 1] * 2.0f - 1.0f;
  float nb2 = nboxes[ri * 4 + 2] * 2.0f - 1.0f;
  float nb3 = nboxes[ri * 4 + 3] * 2.0f - 1.0f;
  float bv = boxb[t] + nb0 * boxw[t * 4 + 0] + nb1 * boxw[t * 4 + 1] +
             nb2 * boxw[t * 4 + 2] + nb3 * boxw[t * 4 + 3];
  red[t] = bv; __syncthreads();
  for (int st = 128; st; st >>= 1) { if (t < st) red[t] += red[t + st]; __syncthreads(); }
  float mu2 = red[0] * (1.0f / 256.0f); __syncthreads();
  float d2 = bv - mu2;
  red[t] = d2 * d2; __syncthreads();
  for (int st = 128; st; st >>= 1) { if (t < st) red[t] += red[t + st]; __syncthreads(); }
  float rstd2 = rsqrtf(red[0] * (1.0f / 256.0f) + 1e-5f);
  frow[1024 + t] = (_Float16)((bv - mu2) * rstd2 * ln2g[t] + ln2b[t]);
}

// ---------------------------------------------------------------------------
// K6: u/v GEMM (g1 factorization). M=512, K=1280, N=2560 combined.
// grid (8, 40): y<20 -> u columns (g1_w[:, :1280]), else v (g1_w[:, 1280:]).
// No bias/ReLU here (applied when building h1). Block 256.
// ---------------------------------------------------------------------------
__global__ void __launch_bounds__(256) k_uv(const _Float16* __restrict__ feats,
                                            const _Float16* __restrict__ g1w,
                                            float* __restrict__ u,
                                            float* __restrict__ v) {
  const int m0 = blockIdx.x * 64;
  const int bn = blockIdx.y;
  const bool isv = bn >= 20;
  const int nc0 = (bn % 20) * 64;
  const int koff = isv ? 1280 : 0;
  const int w = threadIdx.x >> 5, msub = w & 3, nsb = (w >> 2) * 2;
  const _Float16* A = feats + (size_t)(m0 + msub * 16) * 1280;
  const _Float16* B0 = g1w + (size_t)(nc0 + nsb * 16) * 2560 + koff;
  const _Float16* B1 = g1w + (size_t)(nc0 + nsb * 16 + 16) * 2560 + koff;
  v8f acc0 = {}, acc1 = {};
  for (int k = 0; k < 1280; k += 32) {
    v16h a = load_a_frag(A + k, 1280);
    v16h b0 = load_b_frag(B0 + k, 2560);
    v16h b1 = load_b_frag(B1 + k, 2560);
    acc0 = wmma16(a, b0, acc0);
    acc1 = wmma16(a, b1, acc1);
  }
  float* O = (isv ? v : u) + (size_t)(m0 + msub * 16) * 1280 + nc0;
  const int l = threadIdx.x & 31, n = l & 15, hi = l >> 4;
  for (int r = 0; r < 8; ++r) {
    int mm = r + hi * 8;
    O[(size_t)mm * 1280 + nsb * 16 + n] = acc0[r];
    O[(size_t)mm * 1280 + nsb * 16 + 16 + n] = acc1[r];
  }
}

// ---------------------------------------------------------------------------
// K7: fused relation network. Per workgroup: 16 pair-rows of one image.
//   h1 = ReLU(u[b]+v[a]+g1_b)  (built in LDS, f16, 16x1280)
//   h2 = ReLU(h1 @ g2_w^T + g2_b)  (WMMA, stored to LDS f16)
//   h3 = ReLU(h2 @ g3_w^T + g3_b)  (WMMA), column-sum over 16 rows,
//   atomicAdd into out[img][1280].
// grid (256, 8), block 256 (8 waves x 10 n-tiles), dyn LDS = 2*16*1280*2 B.
// ---------------------------------------------------------------------------
__global__ void __launch_bounds__(256) k_relation(const float* __restrict__ u,
                                                  const float* __restrict__ v,
                                                  const float* __restrict__ g1b,
                                                  const _Float16* __restrict__ g2w,
                                                  const float* __restrict__ g2b,
                                                  const _Float16* __restrict__ g3w,
                                                  const float* __restrict__ g3b,
                                                  float* __restrict__ out) {
  extern __shared__ _Float16 sm[];
  _Float16* h1 = sm;               // 16*1280
  _Float16* h2 = sm + 16 * 1280;   // 16*1280
  const int img = blockIdx.y;
  const int row0 = blockIdx.x * 16;
  const int t = threadIdx.x;

  for (int idx = t; idx < 16 * 1280; idx += 256) {
    int m = idx / 1280, k = idx - m * 1280;
    int p = row0 + m;
    int a = p >> 6, b = p & 63;
    float val = u[((size_t)img * 64 + b) * 1280 + k] +
                v[((size_t)img * 64 + a) * 1280 + k] + g1b[k];
    h1[idx] = (_Float16)fmaxf(val, 0.0f);
  }
  __syncthreads();

  const int w = t >> 5, l = t & 31, n = l & 15, hi = l >> 4;
  v8f acc[10] = {};
  for (int k = 0; k < 1280; k += 32) {
    v16h a = load_a_frag(h1 + k, 1280);
    __builtin_prefetch(g2w + (size_t)(w * 160) * 1280 + k + 256, 0, 1);
    for (int j = 0; j < 10; ++j) {
      int nc = (w * 10 + j) * 16;
      v16h b = load_b_frag(g2w + (size_t)nc * 1280 + k, 1280);
      acc[j] = wmma16(a, b, acc[j]);
    }
  }
  for (int j = 0; j < 10; ++j) {
    int nc = (w * 10 + j) * 16 + n;
    float bb = g2b[nc];
    for (int r = 0; r < 8; ++r) {
      int m = r + hi * 8;
      h2[m * 1280 + nc] = (_Float16)fmaxf(acc[j][r] + bb, 0.0f);
    }
  }
  __syncthreads();

  for (int j = 0; j < 10; ++j) acc[j] = (v8f){};
  for (int k = 0; k < 1280; k += 32) {
    v16h a = load_a_frag(h2 + k, 1280);
    __builtin_prefetch(g3w + (size_t)(w * 160) * 1280 + k + 256, 0, 1);
    for (int j = 0; j < 10; ++j) {
      int nc = (w * 10 + j) * 16;
      v16h b = load_b_frag(g3w + (size_t)nc * 1280 + k, 1280);
      acc[j] = wmma16(a, b, acc[j]);
    }
  }
  for (int j = 0; j < 10; ++j) {
    int nc = (w * 10 + j) * 16 + n;
    float bb = g3b[nc];
    float ps = 0.0f;
    for (int r = 0; r < 8; ++r) ps += fmaxf(acc[j][r] + bb, 0.0f);
    ps += __shfl_xor(ps, 16, 32);   // combine the two M-halves of the wave
    if (hi == 0) atomicAdd(out + (size_t)img * 1280 + nc, ps);
  }
}

// ---------------------------------------------------------------------------
// Host launcher
// ---------------------------------------------------------------------------
extern "C" void kernel_launch(void* const* d_in, const int* in_sizes, int n_in,
                              void* d_out, int out_size, void* d_ws, size_t ws_size,
                              hipStream_t stream) {
  (void)in_sizes; (void)n_in; (void)ws_size;
  const float* im_feat = (const float*)d_in[0];
  const float* boxes   = (const float*)d_in[1];
  const float* nboxes  = (const float*)d_in[2];
  const float* proj_w  = (const float*)d_in[3];
  const float* proj_b  = (const float*)d_in[4];
  const float* conv_w  = (const float*)d_in[5];
  const float* conv_b  = (const float*)d_in[6];
  const float* bn_g    = (const float*)d_in[7];
  const float* bn_b    = (const float*)d_in[8];
  const float* bn_m    = (const float*)d_in[9];
  const float* bn_v    = (const float*)d_in[10];
  const float* fc_w    = (const float*)d_in[11];
  const float* fc_b    = (const float*)d_in[12];
  const float* ln1_g   = (const float*)d_in[13];
  const float* ln1_b   = (const float*)d_in[14];
  const float* box_w   = (const float*)d_in[15];
  const float* box_b   = (const float*)d_in[16];
  const float* ln2_g   = (const float*)d_in[17];
  const float* ln2_b   = (const float*)d_in[18];
  const float* g1_w    = (const float*)d_in[19];
  const float* g1_b    = (const float*)d_in[20];
  const float* g2_w    = (const float*)d_in[21];
  const float* g2_b    = (const float*)d_in[22];
  const float* g3_w    = (const float*)d_in[23];
  const float* g3_b    = (const float*)d_in[24];

  char* ws = (char*)d_ws;
  size_t off = 0;
  auto take = [&](size_t bytes) -> char* {
    char* p = ws + off;
    off += (bytes + 255) & ~(size_t)255;
    return p;
  };
  float*     xbuf    = (float*)take(8ull * 1024 * 256 * 4);        // proj out f32
  _Float16*  projw16 = (_Float16*)take(1024ull * 2048 * 2);
  _Float16*  imf16   = (_Float16*)take(8ull * 256 * 2048 * 2);
  _Float16*  rbuf    = (_Float16*)take(512ull * 81 * 1024 * 2);    // padded ROI NHWC f16
  _Float16*  convw16 = (_Float16*)take(256ull * 9216 * 2);
  _Float16*  cout16  = (_Float16*)take(512ull * 49 * 256 * 2);     // conv out f16
  _Float16*  fcw16   = (_Float16*)take(1024ull * 12544 * 2);
  float*     fcout   = (float*)take(512ull * 1024 * 4);
  _Float16*  feats16 = (_Float16*)take(512ull * 1280 * 2);
  _Float16*  g1w16   = (_Float16*)take(1280ull * 2560 * 2);
  _Float16*  g2w16   = (_Float16*)take(1280ull * 1280 * 2);
  _Float16*  g3w16   = (_Float16*)take(1280ull * 1280 * 2);
  float*     ubuf    = (float*)take(512ull * 1280 * 4);
  float*     vbuf    = (float*)take(512ull * 1280 * 4);

  (void)hipMemsetAsync(d_out, 0, (size_t)out_size * sizeof(float), stream);

  const int T = 256;
  k_f32_to_f16<<<(1024 * 2048 + T - 1) / T, T, 0, stream>>>(proj_w, projw16, 1024 * 2048);
  k_f32_to_f16<<<(1280 * 2560 + T - 1) / T, T, 0, stream>>>(g1_w, g1w16, 1280 * 2560);
  k_f32_to_f16<<<(1280 * 1280 + T - 1) / T, T, 0, stream>>>(g2_w, g2w16, 1280 * 1280);
  k_f32_to_f16<<<(1280 * 1280 + T - 1) / T, T, 0, stream>>>(g3_w, g3w16, 1280 * 1280);
  k_cvt_convw<<<(256 * 9216 + T - 1) / T, T, 0, stream>>>(conv_w, convw16);
  k_cvt_fcw<<<(1024 * 12544 + T - 1) / T, T, 0, stream>>>(fc_w, fcw16);
  k_cvt_imf<<<(8 * 256 * 2048 + T - 1) / T, T, 0, stream>>>(im_feat, imf16);
  k_zero_border<<<dim3(512, 32), 128, 0, stream>>>(rbuf);

  k_proj<<<dim3(16, 4, 8), 256, 0, stream>>>(projw16, imf16, proj_b, xbuf);
  k_roialign<<<dim3(512, 49), 256, 0, stream>>>(xbuf, boxes, rbuf);
  k_conv3<<<512, 256, 0, stream>>>(rbuf, convw16, conv_b, bn_g, bn_b, bn_m, bn_v, cout16);
  k_fc<<<dim3(8, 16), 256, 0, stream>>>(cout16, fcw16, fc_b, fcout);
  k_feats<<<512, 256, 0, stream>>>(fcout, ln1_g, ln1_b, nboxes, box_w, box_b,
                                   ln2_g, ln2_b, feats16);
  k_uv<<<dim3(8, 40), 256, 0, stream>>>(feats16, g1w16, ubuf, vbuf);
  k_relation<<<dim3(256, 8), 256, 2 * 16 * 1280 * sizeof(_Float16), stream>>>(
      ubuf, vbuf, g1_b, g2w16, g2_b, g3w16, g3_b, (float*)d_out);
}
